// TriangleAttention_11605001634120
// MI455X (gfx1250) — compile-verified, matrix operand order checked by reading, not run on hardware
//
#include <hip/hip_runtime.h>
#include <hip/hip_bf16.h>

typedef __attribute__((ext_vector_type(16))) _Float16 v16h;
typedef __attribute__((ext_vector_type(8)))  _Float16 v8h;
typedef __attribute__((ext_vector_type(4)))  _Float16 v4h;
typedef __attribute__((ext_vector_type(8)))  float    v8f;

constexpr int kN  = 384;
constexpr int kNN = kN * kN;      // 147456
constexpr int kCZ = 128;
constexpr int kH  = 4;
constexpr int kC  = 32;
constexpr int kHC = kH * kC;      // 128

// ---------------------------------------------------------------------------
// 16x16x32 f16 WMMA wrapper (CDNA5, wave32). D = A(16x32) * B(32x16) + C.
// ---------------------------------------------------------------------------
__device__ __forceinline__ v8f wmma_f16(v16h a, v16h b, v8f c) {
  return __builtin_amdgcn_wmma_f32_16x16x32_f16(false, a, false, b,
                                                (short)0, c, false, false);
}

// ---------------------------------------------------------------------------
// 16x32 fp16 fragment load (A-layout, or B-layout of a transposed operand):
// lane L holds row m = L%16, halves [8*(L/16), +8) and [16+8*(L/16), +8).
// ---------------------------------------------------------------------------
__device__ __forceinline__ v16h gfrag(const _Float16* __restrict__ base,
                                      int strideH, int lane) {
  const _Float16* p = base + (long)(lane & 15) * strideH + ((lane >> 4) << 3);
  v8h lo = *(const v8h*)p;
  v8h hi = *(const v8h*)(p + 16);
  v16h r;
#pragma unroll
  for (int e = 0; e < 8; ++e) { r[e] = lo[e]; r[8 + e] = hi[e]; }
  return r;
}

// Fragment from an LDS tile row (32-half rows): two ds_load_b128.
__device__ __forceinline__ v16h lfrag(const _Float16* p0) {
  v8h lo = *(const v8h*)p0;
  v8h hi = *(const v8h*)(p0 + 16);
  v16h r;
#pragma unroll
  for (int e = 0; e < 8; ++e) { r[e] = lo[e]; r[8 + e] = hi[e]; }
  return r;
}

// ---------------------------------------------------------------------------
// One 16-byte global -> LDS async DMA (CDNA5, tracked by ASYNCcnt).
// VDST carries the LDS byte address.
// A generic pointer into LDS holds the LDS offset in its low 32 bits on
// AMDGPU, so truncating the pointer gives the right VDST value.
// ---------------------------------------------------------------------------
__device__ __forceinline__ void copy16(const _Float16* gsrc, _Float16* ldst) {
#if defined(__gfx1250__)
  asm volatile("global_load_async_to_lds_b128 %0, %1, off"
               :
               : "v"((unsigned)(unsigned long long)ldst), "v"(gsrc)
               : "memory");
#else
  *(v8h*)ldst = *(const v8h*)gsrc;
#endif
}

__device__ __forceinline__ void wait_async_copies() {
#if defined(__gfx1250__)
  asm volatile("s_wait_asynccnt 0x0" ::: "memory");
#endif
}

// ---------------------------------------------------------------------------
// Kernel 1: LayerNorm over c_z (one wave per (i,j) row) + pair-bias projection
//   zn  : fp16 (row, c) layout, NN x 128
//   bias: fp32 [h][q][k] = [h*NN + i*N + j]
// ---------------------------------------------------------------------------
__global__ __launch_bounds__(256) void ln_bias_kernel(
    const float* __restrict__ z, const float* __restrict__ g,
    const float* __restrict__ b, const float* __restrict__ wb,
    _Float16* __restrict__ zn, float* __restrict__ bias) {
  const int lane = threadIdx.x & 31;
  const int wave = threadIdx.x >> 5;
  const long row = (long)blockIdx.x * 8 + wave;

  const float4 zv = ((const float4*)(z + row * kCZ))[lane];  // c = 4*lane..
  float s  = zv.x + zv.y + zv.z + zv.w;
  float s2 = zv.x * zv.x + zv.y * zv.y + zv.z * zv.z + zv.w * zv.w;
#pragma unroll
  for (int m = 1; m < 32; m <<= 1) {
    s  += __shfl_xor(s,  m, 32);
    s2 += __shfl_xor(s2, m, 32);
  }
  const float mu = s * (1.0f / kCZ);
  const float rs = rsqrtf(s2 * (1.0f / kCZ) - mu * mu + 1e-5f);

  const float4 gv = ((const float4*)g)[lane];
  const float4 bv = ((const float4*)b)[lane];
  float zl[4];
  zl[0] = (zv.x - mu) * rs * gv.x + bv.x;
  zl[1] = (zv.y - mu) * rs * gv.y + bv.y;
  zl[2] = (zv.z - mu) * rs * gv.z + bv.z;
  zl[3] = (zv.w - mu) * rs * gv.w + bv.w;

  v4h hv = { (_Float16)zl[0], (_Float16)zl[1], (_Float16)zl[2], (_Float16)zl[3] };
  ((v4h*)(zn + row * kCZ))[lane] = hv;

  // bias_h = sum_c zn[c] * wb[c,h]
  float acc[kH] = {0.f, 0.f, 0.f, 0.f};
  const int c0 = lane * 4;
#pragma unroll
  for (int kk = 0; kk < 4; ++kk)
#pragma unroll
    for (int h = 0; h < kH; ++h) acc[h] += zl[kk] * wb[(c0 + kk) * kH + h];
#pragma unroll
  for (int h = 0; h < kH; ++h)
#pragma unroll
    for (int m = 1; m < 32; m <<= 1) acc[h] += __shfl_xor(acc[h], m, 32);
  if (lane == 0) {
#pragma unroll
    for (int h = 0; h < kH; ++h) bias[(long)h * kNN + row] = acc[h];
  }
}

// ---------------------------------------------------------------------------
// Kernel 2: projection GEMM  Y(M=NN x 128) = zn(fp16) @ W(128x128 fp32->fp16)
// Block = 256 threads (8 waves). Block tile: 128 rows x 64 cols.
// mode: 0=q (scaled, (i,h,j,c)), 1=k ((i,h,j,c)), 2=v (transposed (i,h,c,j)),
//       3=g (sigmoid(x+bg), (i,j,h*C+c))
// ---------------------------------------------------------------------------
__global__ __launch_bounds__(256) void proj_kernel(
    const _Float16* __restrict__ zn, const float* __restrict__ W,
    const float* __restrict__ bg, _Float16* __restrict__ out, int mode) {
  __shared__ _Float16 Wt[64][kCZ];  // [n][k], 16 KB
  const int lane = threadIdx.x & 31;
  const int wave = threadIdx.x >> 5;
  const int colbase = blockIdx.y * 64;
  const int rowbase = blockIdx.x * 128 + wave * 16;

  for (int idx = threadIdx.x; idx < 64 * kCZ; idx += 256) {
    const int n = idx & 63, kk = idx >> 6;
    Wt[n][kk] = (_Float16)W[kk * kHC + colbase + n];
  }
  __syncthreads();

  const v8f zero = {0.f, 0.f, 0.f, 0.f, 0.f, 0.f, 0.f, 0.f};
  v8f acc[4] = {zero, zero, zero, zero};

#pragma unroll
  for (int kk = 0; kk < kCZ; kk += 32) {
    const v16h a = gfrag(zn + (long)rowbase * kCZ + kk, kCZ, lane);
#pragma unroll
    for (int t = 0; t < 4; ++t) {
      const v16h bf = lfrag(&Wt[t * 16 + (lane & 15)][kk + ((lane >> 4) << 3)]);
      acc[t] = wmma_f16(a, bf, acc[t]);
    }
  }

#pragma unroll
  for (int t = 0; t < 4; ++t)
#pragma unroll
    for (int r = 0; r < 8; ++r) {
      const int m   = rowbase + ((lane >> 4) << 3) + r;  // global row (i*N+j)
      const int col = colbase + t * 16 + (lane & 15);    // 0..127 = h*C+c
      float val = acc[t][r];
      const int i = m / kN, j = m % kN;
      const int h = col >> 5, c = col & 31;
      if (mode == 0) {
        out[((i * kH + h) * kN + j) * kC + c] = (_Float16)(val * 0.17677669529663687f);
      } else if (mode == 1) {
        out[((i * kH + h) * kN + j) * kC + c] = (_Float16)val;
      } else if (mode == 2) {
        out[((i * kH + h) * kC + c) * kN + j] = (_Float16)val;   // V transposed
      } else {
        val += bg[col];
        out[(long)m * kHC + col] = (_Float16)(1.0f / (1.0f + __expf(-val)));
      }
    }
}

// ---------------------------------------------------------------------------
// Kernel 3: flash attention per (qblock, i, h). Wave owns 16 q rows.
// K/V 32x32 chunks are DMA'd once per block into double-buffered LDS
// (global_load_async_to_lds_b128, ASYNCcnt), overlapping the next chunk's
// copy with this chunk's WMMAs. Per chunk: 2 score WMMAs (+bias), online
// softmax (cross-lane butterflies within lane-groups of 16 matching the D
// layout), P bounced through per-wave LDS (D->A layout), 2 output WMMAs.
// ---------------------------------------------------------------------------
__global__ __launch_bounds__(256) void attn_kernel(
    const _Float16* __restrict__ q, const _Float16* __restrict__ k,
    const _Float16* __restrict__ vt, const float* __restrict__ bias,
    _Float16* __restrict__ ob) {
  __shared__ _Float16 Kt[2][32][32];  // [buf][j-row][c]      4 KB
  __shared__ _Float16 Vs[2][32][32];  // [buf][c-row][j]      4 KB
  __shared__ _Float16 P[8][16][32];   // per-wave P staging   8 KB
  const int tid  = threadIdx.x;
  const int lane = tid & 31;
  const int wave = tid >> 5;
  const int i = blockIdx.y, h = blockIdx.z;
  const int qbase = blockIdx.x * 128 + wave * 16;
  const long head = (long)(i * kH + h);
  const _Float16* qp = q  + head * kN * kC;
  const _Float16* kp = k  + head * kN * kC;
  const _Float16* vp = vt + head * kC * kN;
  const float*    bp = bias + (long)h * kNN;

  const int mbase = ((lane >> 4) << 3);
  const v16h aq = gfrag(qp + (long)qbase * kC, kC, lane);  // Q rows fixed

  // cooperative K/V tile copy: threads 0..127 -> K, 128..255 -> V.
  const int crow = (tid & 127) >> 2;   // 0..31
  const int cchk = (tid & 3) << 3;     // half offset 0,8,16,24

  // prologue: chunk 0 -> buffer 0
  if (tid < 128) copy16(kp + (long)crow * kC + cchk, &Kt[0][crow][cchk]);
  else           copy16(vp + (long)crow * kN + cchk, &Vs[0][crow][cchk]);

  const v8f zero = {0.f, 0.f, 0.f, 0.f, 0.f, 0.f, 0.f, 0.f};
  v8f o0 = zero, o1 = zero;  // c in [0,16) / [16,32)
  float mrow[8], lrow[8];
#pragma unroll
  for (int r = 0; r < 8; ++r) { mrow[r] = -1e30f; lrow[r] = 0.f; }

  for (int n = 0; n < kN / 32; ++n) {
    const int kb  = n * 32;
    const int buf = n & 1;

    // current buffer ready (own ASYNCcnt), block-wide via barrier; the same
    // barrier proves everyone finished reading buf^1 last iteration.
    wait_async_copies();
    __syncthreads();
    if (kb + 32 < kN) {  // overlap next chunk's DMA with this chunk's WMMAs
      if (tid < 128) copy16(kp + (long)(kb + 32 + crow) * kC + cchk,
                            &Kt[buf ^ 1][crow][cchk]);
      else           copy16(vp + (long)crow * kN + kb + 32 + cchk,
                            &Vs[buf ^ 1][crow][cchk]);
    }

    const v16h bk0 = lfrag(&Kt[buf][lane & 15][(lane >> 4) << 3]);
    const v16h bk1 = lfrag(&Kt[buf][16 + (lane & 15)][(lane >> 4) << 3]);
    v8f s0 = wmma_f16(aq, bk0, zero);
    v8f s1 = wmma_f16(aq, bk1, zero);

#pragma unroll
    for (int r = 0; r < 8; ++r) {
      const int qrow = qbase + mbase + r;
      s0[r] += bp[(long)qrow * kN + kb + (lane & 15)];
      s1[r] += bp[(long)qrow * kN + kb + 16 + (lane & 15)];
      float t = fmaxf(s0[r], s1[r]);
#pragma unroll
      for (int x = 1; x < 16; x <<= 1) t = fmaxf(t, __shfl_xor(t, x, 32));
      const float nm = fmaxf(mrow[r], t);
      const float sc = __expf(mrow[r] - nm);
      const float p0 = __expf(s0[r] - nm);
      const float p1 = __expf(s1[r] - nm);
      float rsum = p0 + p1;
#pragma unroll
      for (int x = 1; x < 16; x <<= 1) rsum += __shfl_xor(rsum, x, 32);
      lrow[r] = lrow[r] * sc + rsum;
      mrow[r] = nm;
      o0[r] *= sc;
      o1[r] *= sc;
      P[wave][mbase + r][lane & 15]        = (_Float16)p0;  // D-layout -> LDS
      P[wave][mbase + r][16 + (lane & 15)] = (_Float16)p1;
    }

    // Re-read P in A-fragment layout (wave-private LDS; in-order DS unit).
    const v16h ap = lfrag(&P[wave][lane & 15][(lane >> 4) << 3]);

    const v16h bv0 = lfrag(&Vs[buf][lane & 15][(lane >> 4) << 3]);
    const v16h bv1 = lfrag(&Vs[buf][16 + (lane & 15)][(lane >> 4) << 3]);
    o0 = wmma_f16(ap, bv0, o0);
    o1 = wmma_f16(ap, bv1, o1);
  }

#pragma unroll
  for (int r = 0; r < 8; ++r) {
    const float inv = 1.0f / lrow[r];
    const int j = qbase + mbase + r;
    _Float16* dst = ob + ((long)i * kN + j) * kHC + h * kC;
    dst[lane & 15]        = (_Float16)(o0[r] * inv);
    dst[16 + (lane & 15)] = (_Float16)(o1[r] * inv);
  }
}

// ---------------------------------------------------------------------------
// Kernel 4: out = (g .* o) @ wo + bo   (fp16 A, LDS-staged fp16 wo, fp32 out)
// ---------------------------------------------------------------------------
__global__ __launch_bounds__(256) void out_kernel(
    const _Float16* __restrict__ gb, const _Float16* __restrict__ ob,
    const float* __restrict__ wo, const float* __restrict__ bo,
    float* __restrict__ out) {
  __shared__ _Float16 Wt[64][kCZ];
  const int lane = threadIdx.x & 31;
  const int wave = threadIdx.x >> 5;
  const int colbase = blockIdx.y * 64;
  const int rowbase = blockIdx.x * 128 + wave * 16;

  for (int idx = threadIdx.x; idx < 64 * kCZ; idx += 256) {
    const int n = idx & 63, kk = idx >> 6;
    Wt[n][kk] = (_Float16)wo[kk * kCZ + colbase + n];
  }
  __syncthreads();

  const v8f zero = {0.f, 0.f, 0.f, 0.f, 0.f, 0.f, 0.f, 0.f};
  v8f acc[4] = {zero, zero, zero, zero};

#pragma unroll
  for (int kk = 0; kk < kCZ; kk += 32) {
    const v16h ag = gfrag(gb + (long)rowbase * kHC + kk, kHC, lane);
    const v16h ao = gfrag(ob + (long)rowbase * kHC + kk, kHC, lane);
    v16h a;
#pragma unroll
    for (int e = 0; e < 16; ++e) a[e] = ag[e] * ao[e];
#pragma unroll
    for (int t = 0; t < 4; ++t) {
      const v16h bf = lfrag(&Wt[t * 16 + (lane & 15)][kk + ((lane >> 4) << 3)]);
      acc[t] = wmma_f16(a, bf, acc[t]);
    }
  }

#pragma unroll
  for (int t = 0; t < 4; ++t)
#pragma unroll
    for (int r = 0; r < 8; ++r) {
      const int m   = rowbase + ((lane >> 4) << 3) + r;
      const int col = colbase + t * 16 + (lane & 15);
      out[(long)m * kCZ + col] = acc[t][r] + bo[col];
    }
}

// ---------------------------------------------------------------------------
extern "C" void kernel_launch(void* const* d_in, const int* in_sizes, int n_in,
                              void* d_out, int out_size, void* d_ws, size_t ws_size,
                              hipStream_t stream) {
  const float* z      = (const float*)d_in[0];
  const float* ln_g   = (const float*)d_in[1];
  const float* ln_b   = (const float*)d_in[2];
  const float* w_bias = (const float*)d_in[3];
  const float* wq     = (const float*)d_in[4];
  const float* wk     = (const float*)d_in[5];
  const float* wv     = (const float*)d_in[6];
  const float* wg     = (const float*)d_in[7];
  const float* bg     = (const float*)d_in[8];
  const float* wo     = (const float*)d_in[9];
  const float* bo     = (const float*)d_in[10];
  float* out = (float*)d_out;

  char* ws = (char*)d_ws;
  const size_t S2 = (size_t)kNN * kHC * 2;  // one fp16 NN x 128 buffer
  _Float16* zn = (_Float16*)(ws);
  _Float16* qb = (_Float16*)(ws + 1 * S2);
  _Float16* kb = (_Float16*)(ws + 2 * S2);
  _Float16* vt = (_Float16*)(ws + 3 * S2);
  _Float16* gb = (_Float16*)(ws + 4 * S2);
  _Float16* ob = (_Float16*)(ws + 5 * S2);
  float*  biasb = (float*)(ws + 6 * S2);    // H * NN fp32

  ln_bias_kernel<<<kNN / 8, 256, 0, stream>>>(z, ln_g, ln_b, w_bias, zn, biasb);

  dim3 g2(kNN / 128, 2);
  proj_kernel<<<g2, 256, 0, stream>>>(zn, wq, nullptr, qb, 0);
  proj_kernel<<<g2, 256, 0, stream>>>(zn, wk, nullptr, kb, 1);
  proj_kernel<<<g2, 256, 0, stream>>>(zn, wv, nullptr, vt, 2);
  proj_kernel<<<g2, 256, 0, stream>>>(zn, wg, bg, gb, 3);

  dim3 g3(kN / 128, kN, kH);
  attn_kernel<<<g3, 256, 0, stream>>>(qb, kb, vt, biasb, ob);

  out_kernel<<<g2, 256, 0, stream>>>(gb, ob, wo, bo, out);
}
